// GATConvEncoder_68264210202880
// MI455X (gfx1250) — matrix operand order repaired and checked by reading, not compiled, not run on hardware
//
#include <hip/hip_runtime.h>

#define NN 25000
#define EE 400000
#define ET (NN + EE)      // 425000 edges incl self loops
#define DMm 384
#define DEe 64
#define HHh 8
#define CCc 48
#define LLl 2
#define NBb 41

typedef __attribute__((ext_vector_type(16))) __bf16 v16bf;
typedef __attribute__((ext_vector_type(8)))  float  v8f;

union FRAG { v16bf v; uint4 q[2]; };

__device__ __forceinline__ unsigned short f2bf(float f) {
  unsigned int u = __float_as_uint(f);
  u += 0x7FFFu + ((u >> 16) & 1u);          // round-to-nearest-even
  return (unsigned short)(u >> 16);
}

__device__ __forceinline__ void atomicMaxF(float* addr, float val) {
  unsigned int* ua = (unsigned int*)addr;
  unsigned int old = __hip_atomic_load(ua, __ATOMIC_RELAXED, __HIP_MEMORY_SCOPE_AGENT);
  while (__uint_as_float(old) < val) {
    unsigned int assumed = old;
    old = atomicCAS(ua, assumed, __float_as_uint(val));
    if (old == assumed) break;
  }
}

// Load a fragment as 2x b128 from a base pointer + compile-time element offset.
__device__ __forceinline__ v16bf load_frag_off(const unsigned short* __restrict__ p, int eoff) {
  FRAG f;
  f.q[0] = *reinterpret_cast<const uint4*>(p + eoff);
  f.q[1] = *reinterpret_cast<const uint4*>(p + eoff + 8);
  return f.v;
}
// A fragment: two 16B runs starting 16 bf16 elements apart within the row.
__device__ __forceinline__ v16bf load_a_off(const unsigned short* __restrict__ p, int eoff) {
  FRAG f;
  f.q[0] = *reinterpret_cast<const uint4*>(p + eoff);
  f.q[1] = *reinterpret_cast<const uint4*>(p + eoff + 16);
  return f.v;
}

__device__ __forceinline__ v8f wmma_bf16(v16bf a, v16bf b, v8f c) {
  return __builtin_amdgcn_wmma_f32_16x16x32_bf16(false, a, false, b, (short)0, c,
                                                 false, false);
}

// ---------------- utility kernels ----------------
__global__ void fill_kernel(float* p, float v, int n) {
  int i = blockIdx.x * blockDim.x + threadIdx.x;
  if (i < n) p[i] = v;
}

// convert + shuffle a [Kdim x 384] f32 weight into fragment-major bf16 layout:
// packed[(((ct*KT + kt)*32 + lane)*16) + j] = bf16(W[kt*32 + (lane>>4)*8 + (j&7) + (j>>3)*16][ct*16 + (lane&15)])
__global__ void pack_w_kernel(const float* __restrict__ src, unsigned short* __restrict__ dst,
                              int Kdim) {
  int i = blockIdx.x * blockDim.x + threadIdx.x;
  int total = Kdim * DMm;
  if (i >= total) return;
  int KT = Kdim >> 5;
  int j = i & 15;
  int lane = (i >> 4) & 31;
  int rest = i >> 9;
  int kt = rest % KT;
  int ct = rest / KT;
  int k = kt * 32 + (lane >> 4) * 8 + (j & 7) + (j >> 3) * 16;
  int n = ct * 16 + (lane & 15);
  dst[i] = f2bf(src[(long)k * DMm + n]);
}

// h = lut[x] * sqrt(DM)
__global__ void embed_kernel(const int* __restrict__ x, const float* __restrict__ lut,
                             float* __restrict__ h, unsigned short* __restrict__ hbf) {
  int i = blockIdx.x * blockDim.x + threadIdx.x;
  if (i >= NN * DMm) return;
  int n = i / DMm, c = i % DMm;
  float v = lut[x[n] * DMm + c] * 19.595917942265423f;  // sqrt(384)
  h[i] = v;
  hbf[i] = f2bf(v);
}

// Gaussian basis -> edge embedding ea (bf16 rows 0..E-1) + scatter mean accumulation
__global__ void edge_embed_kernel(const float* __restrict__ edge_attr,
                                  const float* __restrict__ We0,
                                  const float* __restrict__ be0,
                                  const int* __restrict__ ei,
                                  unsigned short* __restrict__ eabf,
                                  float* __restrict__ loopsum,
                                  float* __restrict__ loopcnt) {
  int e = blockIdx.x;             // one edge per block, 64 threads
  int t = threadIdx.x;
  __shared__ float bs[NBb];
  float d = edge_attr[e];
  if (t < NBb) {
    float diff = (d - 0.2f * (float)t) * 5.0f;      // (d - v)/STEP
    bs[t] = __expf(-diff * diff) * (1.0f / 1.12f);
  }
  __syncthreads();
  float acc = be0[t];
#pragma unroll 1
  for (int j = 0; j < NBb; ++j) acc += bs[j] * We0[j * DEe + t];
  eabf[(long)e * DEe + t] = f2bf(acc);
  int tgt = ei[EE + e];
  atomicAdd(&loopsum[(long)tgt * DEe + t], acc);
  if (t == 0) atomicAdd(&loopcnt[tgt], 1.0f);
}

// self-loop attrs: rows E..Et-1 of ea_full
__global__ void loop_attr_kernel(const float* __restrict__ loopsum,
                                 const float* __restrict__ loopcnt,
                                 unsigned short* __restrict__ eabf) {
  int i = blockIdx.x * blockDim.x + threadIdx.x;
  if (i >= NN * DEe) return;
  int n = i / DEe;
  float c = loopcnt[n];
  c = c < 1.0f ? 1.0f : c;
  eabf[(long)(EE + n) * DEe + (i % DEe)] = f2bf(loopsum[i] / c);
}

// Fused xl = h@Wl+bl, xr = h@Wr+br, res = h@Wres+bias_g   (bf16 WMMA, f32 acc)
__global__ void node_gemm_kernel(const unsigned short* __restrict__ hbf,
                                 const unsigned short* __restrict__ Wlb,
                                 const unsigned short* __restrict__ Wrb,
                                 const unsigned short* __restrict__ Wsb,
                                 const float* __restrict__ bl, const float* __restrict__ br,
                                 const float* __restrict__ bg,
                                 float* __restrict__ xl, float* __restrict__ xr,
                                 float* __restrict__ res) {
  int lane = threadIdx.x & 31;
  int wave = threadIdx.x >> 5;
  int r0 = blockIdx.x * 16;
  int ct = blockIdx.y * 8 + wave;         // 0..23 col tiles
  int col = ct * 16 + (lane & 15);
  int row = r0 + (lane & 15);
  row = row < NN ? row : NN - 1;          // clamp: OOB rows never stored

  // base pointers; kt-dependent offsets are compile-time (A: kt*32 elems, B: kt*512 elems)
  const unsigned short* pa = hbf + (long)row * DMm + (lane >> 4) * 8;
  size_t wb = ((size_t)(ct * (DMm / 32)) * 32 + lane) * 16;
  const unsigned short* pl = Wlb + wb;
  const unsigned short* pr = Wrb + wb;
  const unsigned short* ps = Wsb + wb;

  v8f aL = {}, aR = {}, aS = {};
#pragma unroll
  for (int kt = 0; kt < DMm / 32; ++kt) {
    v16bf a = load_a_off(pa, kt * 32);
    aL = wmma_bf16(a, load_frag_off(pl, kt * 512), aL);
    aR = wmma_bf16(a, load_frag_off(pr, kt * 512), aR);
    aS = wmma_bf16(a, load_frag_off(ps, kt * 512), aS);
  }
  float vbl = bl[col], vbr = br[col], vbg = bg[col];
  if (r0 + 16 <= NN) {                    // fast path: full tile
#pragma unroll
    for (int v = 0; v < 8; ++v) {
      long o = (long)(r0 + v + 8 * (lane >> 4)) * DMm + col;
      xl[o] = aL[v] + vbl;
      xr[o] = aR[v] + vbr;
      res[o] = aS[v] + vbg;
    }
  } else {
#pragma unroll
    for (int v = 0; v < 8; ++v) {
      int rr = r0 + v + 8 * (lane >> 4);
      if (rr < NN) {
        long o = (long)rr * DMm + col;
        xl[o] = aL[v] + vbl;
        xr[o] = aR[v] + vbr;
        res[o] = aS[v] + vbg;
      }
    }
  }
}

// Fused e_proj GEMM (ea_full @ We) + m = leakyrelu(xl[src]+xr[tgt]+eproj),
// alpha logit = sum_c m*att, atomic segment-max. 16 edges per block, 8 waves x 3 tiles.
__global__ void edge_alpha_kernel(const unsigned short* __restrict__ eabf,
                                  const unsigned short* __restrict__ Webf,  // packed, KT=2
                                  const float* __restrict__ xl, const float* __restrict__ xr,
                                  const float* __restrict__ att,
                                  const int* __restrict__ ei,
                                  float* __restrict__ alpha, float* __restrict__ segmax) {
  __shared__ float alds[16 * HHh];
  __shared__ int s_src[16], s_tgt[16];
  int tid = threadIdx.x;
  int e0 = blockIdx.x * 16;
  if (tid < 16) {
    int e = e0 + tid, s = 0, t = 0;
    if (e < ET) {
      if (e < EE) { s = ei[e]; t = ei[EE + e]; }
      else        { s = e - EE; t = e - EE; }
    }
    s_src[tid] = s; s_tgt[tid] = t;
  }
  if (tid < 16 * HHh) alds[tid] = 0.0f;
  __syncthreads();

  int lane = tid & 31, wave = tid >> 5;
  int erow = e0 + (lane & 15);
  erow = erow < ET ? erow : ET - 1;       // clamp: OOB edges never accumulated
  // A fragments are invariant across col tiles: load once (2x b128 each)
  const unsigned short* pa = eabf + (long)erow * DEe + (lane >> 4) * 8;
  v16bf a0 = load_a_off(pa, 0);
  v16bf a1 = load_a_off(pa, 32);
  const unsigned short* pb0 = Webf + ((size_t)(wave * 3 * 2) * 32 + lane) * 16;
#pragma unroll
  for (int t3 = 0; t3 < 3; ++t3) {
    int ct = wave * 3 + t3;
    int col = ct * 16 + (lane & 15);
    v8f acc = {};
    acc = wmma_bf16(a0, load_frag_off(pb0, t3 * 1024), acc);
    acc = wmma_bf16(a1, load_frag_off(pb0, t3 * 1024 + 512), acc);
    float av = att[col];
    int hd = col / CCc;
#pragma unroll
    for (int v = 0; v < 8; ++v) {
      int r = v + 8 * (lane >> 4);
      int ee = e0 + r;
      if (ee < ET) {
        int s = s_src[r], t = s_tgt[r];
        float m = acc[v] + xl[(long)s * DMm + col] + xr[(long)t * DMm + col];
        m = m > 0.0f ? m : 0.2f * m;   // leaky relu 0.2
        atomicAdd(&alds[r * HHh + hd], m * av);
      }
    }
  }
  __syncthreads();
  if (tid < 16 * HHh) {
    int r = tid >> 3, hd = tid & 7;
    int ee = e0 + r;
    if (ee < ET) {
      float a = alds[tid];
      alpha[(long)ee * HHh + hd] = a;
      atomicMaxF(&segmax[(long)s_tgt[r] * HHh + hd], a);
    }
  }
}

// w = exp(alpha - segmax[tgt]); segsum += w
__global__ void edge_exp_kernel(float* __restrict__ alpha, const float* __restrict__ segmax,
                                float* __restrict__ segsum, const int* __restrict__ ei) {
  int i = blockIdx.x * blockDim.x + threadIdx.x;
  if (i >= ET * HHh) return;
  int e = i >> 3, hd = i & 7;
  int t = (e < EE) ? ei[EE + e] : (e - EE);
  float w = __expf(alpha[i] - segmax[(long)t * HHh + hd]);
  alpha[i] = w;
  atomicAdd(&segsum[(long)t * HHh + hd], w);
}

// accum[tgt] += xl[src] * w/(segsum[tgt]+1e-16)
__global__ void edge_scatter_kernel(const float* __restrict__ alpha,
                                    const float* __restrict__ segsum,
                                    const float* __restrict__ xl,
                                    float* __restrict__ accum,
                                    const int* __restrict__ ei) {
  int e = blockIdx.x;            // 128 threads
  __shared__ int sst[2];
  __shared__ float coef[HHh];
  if (threadIdx.x == 0) {
    sst[0] = (e < EE) ? ei[e] : e - EE;
    sst[1] = (e < EE) ? ei[EE + e] : e - EE;
  }
  __syncthreads();
  int ss = sst[0], st = sst[1];
  if (threadIdx.x < HHh)
    coef[threadIdx.x] = alpha[(long)e * HHh + threadIdx.x] /
                        (segsum[(long)st * HHh + threadIdx.x] + 1e-16f);
  __syncthreads();
#pragma unroll
  for (int it = 0; it < 3; ++it) {
    int n = it * 128 + threadIdx.x;
    atomicAdd(&accum[(long)st * DMm + n], xl[(long)ss * DMm + n] * coef[n / CCc]);
  }
}

// out = accum + res; LayerNorm -> h (f32) and hbf (bf16)
__global__ void ln_kernel(const float* __restrict__ accum, const float* __restrict__ res,
                          const float* __restrict__ lng, const float* __restrict__ lnb,
                          float* __restrict__ h, unsigned short* __restrict__ hbf) {
  int n = blockIdx.x;
  int tid = threadIdx.x;       // 128
  __shared__ float red[128];
  long b = (long)n * DMm;
  float v0 = accum[b + tid] + res[b + tid];
  float v1 = accum[b + tid + 128] + res[b + tid + 128];
  float v2 = accum[b + tid + 256] + res[b + tid + 256];
  red[tid] = v0 + v1 + v2;
  __syncthreads();
  for (int o = 64; o > 0; o >>= 1) { if (tid < o) red[tid] += red[tid + o]; __syncthreads(); }
  float mu = red[0] * (1.0f / DMm);
  __syncthreads();
  float d0 = v0 - mu, d1 = v1 - mu, d2 = v2 - mu;
  red[tid] = d0 * d0 + d1 * d1 + d2 * d2;
  __syncthreads();
  for (int o = 64; o > 0; o >>= 1) { if (tid < o) red[tid] += red[tid + o]; __syncthreads(); }
  float rsig = rsqrtf(red[0] * (1.0f / DMm) + 1e-5f);
  float o0 = d0 * rsig * lng[tid] + lnb[tid];
  float o1 = d1 * rsig * lng[tid + 128] + lnb[tid + 128];
  float o2 = d2 * rsig * lng[tid + 256] + lnb[tid + 256];
  h[b + tid] = o0;           hbf[b + tid] = f2bf(o0);
  h[b + tid + 128] = o1;     hbf[b + tid + 128] = f2bf(o1);
  h[b + tid + 256] = o2;     hbf[b + tid + 256] = f2bf(o2);
}

// BatchNorm: per-column stats over nodes
__global__ void bn_stats_kernel(const float* __restrict__ h, float* __restrict__ stats) {
  int c = blockIdx.x;          // 384 blocks, 256 threads
  int tid = threadIdx.x;
  float s = 0.0f, s2 = 0.0f;
  for (int n = tid; n < NN; n += 256) {
    float v = h[(long)n * DMm + c];
    s += v; s2 += v * v;
  }
  __shared__ float r1[256], r2[256];
  r1[tid] = s; r2[tid] = s2;
  __syncthreads();
  for (int o = 128; o > 0; o >>= 1) {
    if (tid < o) { r1[tid] += r1[tid + o]; r2[tid] += r2[tid + o]; }
    __syncthreads();
  }
  if (tid == 0) {
    float mu = r1[0] / (float)NN;
    float var = r2[0] / (float)NN - mu * mu;
    stats[c] = mu;
    stats[DMm + c] = rsqrtf(var + 1e-5f);
  }
}

__global__ void bn_apply_kernel(const float* __restrict__ h, const float* __restrict__ stats,
                                const float* __restrict__ g, const float* __restrict__ bb,
                                float* __restrict__ out) {
  int i = blockIdx.x * blockDim.x + threadIdx.x;
  if (i >= NN * DMm) return;
  int c = i % DMm;
  out[i] = (h[i] - stats[c]) * stats[DMm + c] * g[c] + bb[c];
}

// ---------------- launch ----------------
extern "C" void kernel_launch(void* const* d_in, const int* in_sizes, int n_in,
                              void* d_out, int out_size, void* d_ws, size_t ws_size,
                              hipStream_t stream) {
  const int*   x        = (const int*)d_in[0];
  const int*   ei       = (const int*)d_in[1];
  const float* eattr    = (const float*)d_in[2];
  const float* lut      = (const float*)d_in[3];
  const float* We0      = (const float*)d_in[4];
  const float* be0      = (const float*)d_in[5];
  const float* Wl       = (const float*)d_in[6];
  const float* bl       = (const float*)d_in[7];
  const float* Wr       = (const float*)d_in[8];
  const float* br       = (const float*)d_in[9];
  const float* We       = (const float*)d_in[10];
  const float* att      = (const float*)d_in[11];
  const float* Wres     = (const float*)d_in[12];
  const float* bias_g   = (const float*)d_in[13];
  const float* ln_g     = (const float*)d_in[14];
  const float* ln_b     = (const float*)d_in[15];
  const float* bn_g     = (const float*)d_in[16];
  const float* bn_b     = (const float*)d_in[17];
  float* out = (float*)d_out;

  char* ws = (char*)d_ws;
  size_t off = 0;
  auto alloc = [&](size_t bytes) { size_t o = off; off += (bytes + 255) & ~(size_t)255; return o; };
  float* h        = (float*)(ws + alloc((size_t)NN * DMm * 4));
  float* res      = (float*)(ws + alloc((size_t)NN * DMm * 4));
  float* xl       = (float*)(ws + alloc((size_t)NN * DMm * 4));
  float* xr       = (float*)(ws + alloc((size_t)NN * DMm * 4));
  float* accum    = (float*)(ws + alloc((size_t)NN * DMm * 4));
  unsigned short* hbf   = (unsigned short*)(ws + alloc((size_t)NN * DMm * 2));
  unsigned short* wlbf  = (unsigned short*)(ws + alloc((size_t)LLl * DMm * DMm * 2));
  unsigned short* wrbf  = (unsigned short*)(ws + alloc((size_t)LLl * DMm * DMm * 2));
  unsigned short* wsbf  = (unsigned short*)(ws + alloc((size_t)LLl * DMm * DMm * 2));
  unsigned short* webf  = (unsigned short*)(ws + alloc((size_t)LLl * DEe * DMm * 2));
  unsigned short* eabf  = (unsigned short*)(ws + alloc((size_t)ET * DEe * 2));
  float* alpha    = (float*)(ws + alloc((size_t)ET * HHh * 4));
  float* segmax   = (float*)(ws + alloc((size_t)NN * HHh * 4));
  float* segsum   = (float*)(ws + alloc((size_t)NN * HHh * 4));
  float* loopsum  = (float*)(ws + alloc((size_t)NN * DEe * 4));
  float* loopcnt  = (float*)(ws + alloc((size_t)NN * 4));
  float* bnstats  = (float*)(ws + alloc((size_t)2 * DMm * 4));

  auto cdiv = [](long a, long b) { return (int)((a + b - 1) / b); };

  // weight conversion + fragment-major packing (bf16)
  for (int l = 0; l < LLl; ++l) {
    pack_w_kernel<<<cdiv(DMm * DMm, 256), 256, 0, stream>>>(
        Wl + (size_t)l * DMm * DMm, wlbf + (size_t)l * DMm * DMm, DMm);
    pack_w_kernel<<<cdiv(DMm * DMm, 256), 256, 0, stream>>>(
        Wr + (size_t)l * DMm * DMm, wrbf + (size_t)l * DMm * DMm, DMm);
    pack_w_kernel<<<cdiv(DMm * DMm, 256), 256, 0, stream>>>(
        Wres + (size_t)l * DMm * DMm, wsbf + (size_t)l * DMm * DMm, DMm);
    pack_w_kernel<<<cdiv(DEe * DMm, 256), 256, 0, stream>>>(
        We + (size_t)l * DEe * DMm, webf + (size_t)l * DEe * DMm, DEe);
  }

  // embedding
  embed_kernel<<<cdiv((long)NN * DMm, 256), 256, 0, stream>>>(x, lut, h, hbf);

  // edge embedding + self-loop mean
  fill_kernel<<<cdiv((long)NN * DEe, 256), 256, 0, stream>>>(loopsum, 0.0f, NN * DEe);
  fill_kernel<<<cdiv(NN, 256), 256, 0, stream>>>(loopcnt, 0.0f, NN);
  edge_embed_kernel<<<EE, 64, 0, stream>>>(eattr, We0, be0, ei, eabf, loopsum, loopcnt);
  loop_attr_kernel<<<cdiv((long)NN * DEe, 256), 256, 0, stream>>>(loopsum, loopcnt, eabf);

  for (int l = 0; l < LLl; ++l) {
    const unsigned short* wl_l = wlbf + (size_t)l * DMm * DMm;
    const unsigned short* wr_l = wrbf + (size_t)l * DMm * DMm;
    const unsigned short* ws_l = wsbf + (size_t)l * DMm * DMm;
    const unsigned short* we_l = webf + (size_t)l * DEe * DMm;

    fill_kernel<<<cdiv((long)NN * HHh, 256), 256, 0, stream>>>(segmax, -1e30f, NN * HHh);
    fill_kernel<<<cdiv((long)NN * HHh, 256), 256, 0, stream>>>(segsum, 0.0f, NN * HHh);
    fill_kernel<<<cdiv((long)NN * DMm, 256), 256, 0, stream>>>(accum, 0.0f, NN * DMm);

    dim3 g1(cdiv(NN, 16), 3, 1);
    node_gemm_kernel<<<g1, 256, 0, stream>>>(hbf, wl_l, wr_l, ws_l,
                                             bl + l * DMm, br + l * DMm, bias_g + l * DMm,
                                             xl, xr, res);
    edge_alpha_kernel<<<cdiv(ET, 16), 256, 0, stream>>>(eabf, we_l, xl, xr,
                                                        att + l * DMm, ei, alpha, segmax);
    edge_exp_kernel<<<cdiv((long)ET * HHh, 256), 256, 0, stream>>>(alpha, segmax, segsum, ei);
    edge_scatter_kernel<<<ET, 128, 0, stream>>>(alpha, segsum, xl, accum, ei);
    ln_kernel<<<NN, 128, 0, stream>>>(accum, res, ln_g + l * DMm, ln_b + l * DMm, h, hbf);
  }

  bn_stats_kernel<<<DMm, 256, 0, stream>>>(h, bnstats);
  bn_apply_kernel<<<cdiv((long)NN * DMm, 256), 256, 0, stream>>>(h, bnstats, bn_g, bn_b, out);
}
